// Attention_89421219103442
// MI455X (gfx1250) — matrix operand order repaired and checked by reading, not compiled, not run on hardware
//
#include <hip/hip_runtime.h>
#include <hip/hip_bf16.h>

typedef __bf16 bf16;
typedef __attribute__((ext_vector_type(4)))  bf16  v4bf;
typedef __attribute__((ext_vector_type(8)))  bf16  v8bf;
typedef __attribute__((ext_vector_type(16))) bf16  v16bf;
typedef __attribute__((ext_vector_type(8)))  float v8f;

#define BATCH   8
#define SEQ     2048
#define DMODEL  1024
#define DK      512
#define ROWS    (BATCH * SEQ)          // 16384
#define SROW    2056                   // padded LDS score stride (bf16 elems)
#define SCALE   0.04419417382415922f   // 1/sqrt(512)
#define SMEM_BYTES (16 * SROW * 2 + 16 * 16 * 4)  // 66816

// ---------------------------------------------------------------------------
// WMMA fragment helpers (ISA 7.12.2 layouts, wave32)
// ---------------------------------------------------------------------------
__device__ __forceinline__ v16bf frag_cat(v8bf lo, v8bf hi) {
  return __builtin_shufflevector(lo, hi, 0,1,2,3,4,5,6,7,8,9,10,11,12,13,14,15);
}

// A-matrix 16x32 bf16, row-major source, base points at (row0, k0).
// lane m=lane&15, sel=(lane>>4): VGPR0-3 = K[sel*8 .. sel*8+7], VGPR4-7 = +16.
__device__ __forceinline__ v16bf load_a_frag(const bf16* base, int ld, int m16, int sel) {
  const bf16* p = base + (size_t)m16 * ld + sel * 8;
  v8bf lo = *(const v8bf*)(p);
  v8bf hi = *(const v8bf*)(p + 16);
  return frag_cat(lo, hi);
}

// B-matrix 32x16 bf16, n-major source ([n][k]), base points at (n0, k0).
// lane n=lane&15, sel=(lane>>4): VGPR0-3 = K[sel*8 .. sel*8+7], VGPR4-7 = +16.
__device__ __forceinline__ v16bf load_b_frag(const bf16* base, int ld, int n16, int sel) {
  const bf16* p = base + (size_t)n16 * ld + sel * 8;
  v8bf lo = *(const v8bf*)(p);
  v8bf hi = *(const v8bf*)(p + 16);
  return frag_cat(lo, hi);
}

__device__ __forceinline__ v8f wmma_bf16(v16bf a, v16bf b, v8f c) {
  return __builtin_amdgcn_wmma_f32_16x16x32_bf16(false, a, false, b, (short)0, c, false, false);
}

// ---------------------------------------------------------------------------
// Kernel 1: x fp32 -> bf16
// ---------------------------------------------------------------------------
__global__ __launch_bounds__(256)
void attnqkv_cvt_x(const float4* __restrict__ x, v4bf* __restrict__ o, int n4) {
  int i = blockIdx.x * blockDim.x + threadIdx.x;
  if (i >= n4) return;
  float4 v = x[i];
  v4bf r;
  r[0] = (bf16)v.x; r[1] = (bf16)v.y; r[2] = (bf16)v.z; r[3] = (bf16)v.w;
  o[i] = r;
}

// ---------------------------------------------------------------------------
// Kernel 2: W[k][n] fp32 -> Wt[z][n][k] bf16 (n-major = WMMA B layout)
// ---------------------------------------------------------------------------
__global__ __launch_bounds__(256)
void attnqkv_cvt_w(const float* __restrict__ Wq, const float* __restrict__ Wk,
                   const float* __restrict__ Wv, bf16* __restrict__ Wt) {
  int idx = blockIdx.x * blockDim.x + threadIdx.x;
  if (idx >= 3 * DK * DMODEL) return;
  int z  = idx >> 19;                 // / (512*1024)
  int r  = idx & ((DK * DMODEL) - 1);
  int k  = r & (DMODEL - 1);
  int n  = r >> 10;                   // / 1024
  const float* W = (z == 0) ? Wq : (z == 1) ? Wk : Wv;
  Wt[idx] = (bf16)W[(size_t)k * DK + n];
}

// ---------------------------------------------------------------------------
// Kernel 3: QKV projection.  out = x @ W + b  (M=16384, N=512, K=1024)
// grid (256,2,3); block 256 = 8 waves (2 m-groups x 4 n-groups).
// Per-wave tile 32x64 -> 8 f32 accumulators (64 VGPRs): no spills.
// z=0 -> Q row-major bf16; z=1 -> K row-major bf16; z=2 -> V^T [d][s] bf16.
// ---------------------------------------------------------------------------
__global__ __launch_bounds__(256)
void attnqkv_proj(const bf16* __restrict__ xb, const bf16* __restrict__ Wt,
                  const float* __restrict__ bq, const float* __restrict__ bk,
                  const float* __restrict__ bv,
                  bf16* __restrict__ Qw, bf16* __restrict__ Kw, bf16* __restrict__ Vtw) {
  const int z    = blockIdx.z;
  const bf16* W  = Wt + (size_t)z * (DK * DMODEL);
  const float* bias = (z == 0) ? bq : (z == 1) ? bk : bv;

  const int tid  = threadIdx.x;
  const int wave = tid >> 5;
  const int lane = tid & 31;
  const int m16  = lane & 15;
  const int sel  = (lane >> 4) & 1;
  const int wm   = wave & 1;                         // 2 m-groups of 32 rows
  const int wn   = wave >> 1;                        // 4 n-groups of 64 cols
  const int mbase = blockIdx.x * 64 + wm * 32;
  const int nbase = blockIdx.y * 256 + wn * 64;

  v8f acc[2][4] = {};

  for (int kk = 0; kk < DMODEL / 32; ++kk) {
    const int k0 = kk * 32;
    v16bf a0 = load_a_frag(xb + (size_t)mbase * DMODEL + k0, DMODEL, m16, sel);
    v16bf a1 = load_a_frag(xb + (size_t)(mbase + 16) * DMODEL + k0, DMODEL, m16, sel);
#pragma unroll
    for (int t = 0; t < 4; ++t) {
      v16bf bb = load_b_frag(W + (size_t)(nbase + t * 16) * DMODEL + k0, DMODEL, m16, sel);
      acc[0][t] = wmma_bf16(a0, bb, acc[0][t]);
      acc[1][t] = wmma_bf16(a1, bb, acc[1][t]);
    }
  }

#pragma unroll
  for (int t = 0; t < 4; ++t) {
    const int col = nbase + t * 16 + m16;
    const float bv_ = bias[col];
#pragma unroll
    for (int mt = 0; mt < 2; ++mt) {
#pragma unroll
      for (int i = 0; i < 8; ++i) {
        const int row = mbase + mt * 16 + i + sel * 8;   // C layout: VGPR i -> M = i (+8 hi half)
        const float val = acc[mt][t][i] + bv_;
        if (z == 0)      Qw[(size_t)row * DK + col] = (bf16)val;
        else if (z == 1) Kw[(size_t)row * DK + col] = (bf16)val;
        else {
          const int bb_ = row >> 11, s = row & (SEQ - 1);
          Vtw[((size_t)bb_ * DK + col) * SEQ + s] = (bf16)val;   // V^T, contiguous in s
        }
      }
    }
  }
}

// ---------------------------------------------------------------------------
// Kernel 4: attention. 1 block = 1 batch x 16 queries; 8 waves.
// Full 16x2048 score stripe lives in LDS (bf16, padded stride).
// Phase-1 key tiles processed in 4 groups of 4 to keep accums in registers.
// ---------------------------------------------------------------------------
__global__ __launch_bounds__(256)
void attnqkv_attn(const bf16* __restrict__ Qw, const bf16* __restrict__ Kw,
                  const bf16* __restrict__ Vtw, float* __restrict__ out) {
  extern __shared__ char smem_raw[];
  bf16*  S   = (bf16*)smem_raw;                 // [16][SROW]
  float* red = (float*)(smem_raw + 16 * SROW * 2);

  const int tid  = threadIdx.x;
  const int wave = tid >> 5;
  const int lane = tid & 31;
  const int m16  = lane & 15;
  const int sel  = (lane >> 4) & 1;

  const int b  = blockIdx.x >> 7;               // batch
  const int q0 = (blockIdx.x & 127) * 16;       // query tile

  const bf16* Qb  = Qw  + ((size_t)b * SEQ + q0) * DK;
  const bf16* Kb  = Kw  + (size_t)b * SEQ * DK;
  const bf16* Vtb = Vtw + (size_t)b * DK * SEQ;

  // ---- Phase 1: S = (Q K^T) * scale; wave covers 256 keys in 4 groups of 4 tiles ----
  for (int g = 0; g < 4; ++g) {
    const int s0g = wave * 256 + g * 64;
    v8f acc[4] = {};
    for (int kk = 0; kk < DK / 32; ++kk) {
      const int k0 = kk * 32;
      v16bf a = load_a_frag(Qb + k0, DK, m16, sel);
#pragma unroll
      for (int t = 0; t < 4; ++t) {
        v16bf bb = load_b_frag(Kb + (size_t)(s0g + t * 16) * DK + k0, DK, m16, sel);
        acc[t] = wmma_bf16(a, bb, acc[t]);
      }
    }
#pragma unroll
    for (int t = 0; t < 4; ++t) {
      const int c = s0g + t * 16 + m16;
#pragma unroll
      for (int i = 0; i < 8; ++i)
        S[(i + sel * 8) * SROW + c] = (bf16)(acc[t][i] * SCALE);
    }
  }
  __syncthreads();

  // ---- Phase 2: row-wise softmax in LDS (16 threads per row) ----
  {
    const int r = tid >> 4, j = tid & 15;
    float mx = -3.0e38f;
    for (int c = j; c < SEQ; c += 16) mx = fmaxf(mx, (float)S[r * SROW + c]);
    red[r * 16 + j] = mx;
    __syncthreads();
    float rmax = red[r * 16];
#pragma unroll
    for (int t = 1; t < 16; ++t) rmax = fmaxf(rmax, red[r * 16 + t]);
    __syncthreads();
    float sm = 0.f;
    for (int c = j; c < SEQ; c += 16) {
      const int id = r * SROW + c;
      const float p = exp2f(((float)S[id] - rmax) * 1.44269504f);
      S[id] = (bf16)p;
      sm += p;
    }
    red[r * 16 + j] = sm;
    __syncthreads();
    float rs = 0.f;
#pragma unroll
    for (int t = 0; t < 16; ++t) rs += red[r * 16 + t];
    const float inv = 1.0f / rs;
    for (int c = j; c < SEQ; c += 16) {
      const int id = r * SROW + c;
      S[id] = (bf16)((float)S[id] * inv);
    }
  }
  __syncthreads();

  // ---- Phase 3: O = P V; wave covers 64 head-dims = 4 n-tiles ----
  {
    const int d0 = wave * 64;
    v8f acc[4] = {};
    for (int ss = 0; ss < SEQ / 32; ++ss) {
      const int k0 = ss * 32;
      v16bf a = load_a_frag(S + k0, SROW, m16, sel);      // P fragment from LDS
#pragma unroll
      for (int t = 0; t < 4; ++t) {
        v16bf bb = load_b_frag(Vtb + (size_t)(d0 + t * 16) * SEQ + k0, SEQ, m16, sel);
        acc[t] = wmma_bf16(a, bb, acc[t]);
      }
    }
    float* Ob = out + ((size_t)b * SEQ + q0) * DK;
#pragma unroll
    for (int t = 0; t < 4; ++t) {
      const int col = d0 + t * 16 + m16;
#pragma unroll
      for (int i = 0; i < 8; ++i)
        Ob[(size_t)(i + sel * 8) * DK + col] = acc[t][i];
    }
  }
}

// ---------------------------------------------------------------------------
extern "C" void kernel_launch(void* const* d_in, const int* in_sizes, int n_in,
                              void* d_out, int out_size, void* d_ws, size_t ws_size,
                              hipStream_t stream) {
  (void)in_sizes; (void)n_in; (void)out_size; (void)ws_size;
  const float* x  = (const float*)d_in[0];
  const float* Wq = (const float*)d_in[1];
  const float* bq = (const float*)d_in[2];
  const float* Wk = (const float*)d_in[3];
  const float* bk = (const float*)d_in[4];
  const float* Wv = (const float*)d_in[5];
  const float* bv = (const float*)d_in[6];
  float* out = (float*)d_out;

  char* ws = (char*)d_ws;
  bf16* xb  = (bf16*)(ws);                     // 33,554,432 B
  bf16* Wt  = (bf16*)(ws + 33554432);          //  3,145,728 B
  bf16* Qw  = (bf16*)(ws + 36700160);          // 16,777,216 B
  bf16* Kw  = (bf16*)(ws + 53477376);          // 16,777,216 B
  bf16* Vtw = (bf16*)(ws + 70254592);          // 16,777,216 B  (total ~83 MB)

  attnqkv_cvt_x<<<(ROWS * DMODEL / 4 + 255) / 256, 256, 0, stream>>>(
      (const float4*)x, (v4bf*)xb, ROWS * DMODEL / 4);
  attnqkv_cvt_w<<<(3 * DK * DMODEL + 255) / 256, 256, 0, stream>>>(Wq, Wk, Wv, Wt);
  attnqkv_proj<<<dim3(ROWS / 64, 2, 3), 256, 0, stream>>>(xb, Wt, bq, bk, bv, Qw, Kw, Vtw);

  (void)hipFuncSetAttribute((const void*)attnqkv_attn,
                            hipFuncAttributeMaxDynamicSharedMemorySize, SMEM_BYTES);
  attnqkv_attn<<<BATCH * (SEQ / 16), 256, SMEM_BYTES, stream>>>(Qw, Kw, Vtw, out);
}
